// PatchExpanding_16690242912999
// MI455X (gfx1250) — matrix operand order repaired and checked by reading, not compiled, not run on hardware
//
#include <hip/hip_runtime.h>

#define DIM  192
#define CIN  (2 * DIM)   // 384  (K)
#define COUT (8 * DIM)   // 1536 (N, of which 1152 are useful)
#define EPS  1e-5f

typedef __attribute__((ext_vector_type(16))) __bf16 v16bf;
typedef __attribute__((ext_vector_type(8)))  float  v8f;

// ---------------------------------------------------------------------------
// Kernel 1: LayerNorm over last dim (384), emit bf16 row-major A matrix.
// One wave (32 lanes) per row, 12 channels per lane (3 x float4).
// ---------------------------------------------------------------------------
__global__ __launch_bounds__(256) void ln_kernel(const float* __restrict__ x,
                                                 const float* __restrict__ gamma,
                                                 const float* __restrict__ beta,
                                                 __bf16* __restrict__ xn, int M) {
    const int wave = threadIdx.x >> 5;
    const int lane = threadIdx.x & 31;
    const int row  = blockIdx.x * 8 + wave;
    if (row >= M) return;

    const float4* xr4 = (const float4*)(x + (size_t)row * CIN) + lane * 3;
    float4 a0 = xr4[0], a1 = xr4[1], a2 = xr4[2];
    float v[12] = {a0.x, a0.y, a0.z, a0.w, a1.x, a1.y, a1.z, a1.w,
                   a2.x, a2.y, a2.z, a2.w};

    float s = 0.f, s2 = 0.f;
#pragma unroll
    for (int j = 0; j < 12; ++j) { s += v[j]; s2 += v[j] * v[j]; }
#pragma unroll
    for (int off = 16; off > 0; off >>= 1) {
        s  += __shfl_xor(s,  off, 32);
        s2 += __shfl_xor(s2, off, 32);
    }
    const float mu  = s * (1.f / CIN);
    const float var = s2 * (1.f / CIN) - mu * mu;
    const float rs  = rsqrtf(var + EPS);

    const float4* g4 = (const float4*)(gamma) + lane * 3;
    const float4* b4 = (const float4*)(beta)  + lane * 3;
    float4 g0 = g4[0], g1 = g4[1], g2 = g4[2];
    float4 be0 = b4[0], be1 = b4[1], be2 = b4[2];
    float g[12] = {g0.x, g0.y, g0.z, g0.w, g1.x, g1.y, g1.z, g1.w,
                   g2.x, g2.y, g2.z, g2.w};
    float bb[12] = {be0.x, be0.y, be0.z, be0.w, be1.x, be1.y, be1.z, be1.w,
                    be2.x, be2.y, be2.z, be2.w};

    unsigned pk[6];
#pragma unroll
    for (int j = 0; j < 6; ++j) {
        float y0 = (v[2 * j + 0] - mu) * rs * g[2 * j + 0] + bb[2 * j + 0];
        float y1 = (v[2 * j + 1] - mu) * rs * g[2 * j + 1] + bb[2 * j + 1];
        union { __bf16 h[2]; unsigned u; } cv;
        cv.h[0] = (__bf16)y0;
        cv.h[1] = (__bf16)y1;
        pk[j] = cv.u;
    }
    uint2* dst = (uint2*)(xn + (size_t)row * CIN + lane * 12);
    dst[0] = make_uint2(pk[0], pk[1]);
    dst[1] = make_uint2(pk[2], pk[3]);
    dst[2] = make_uint2(pk[4], pk[5]);
}

// ---------------------------------------------------------------------------
// Kernel 2: W (K x N, f32, row-major) -> Wt (N x K, bf16) for B fragments.
// ---------------------------------------------------------------------------
__global__ __launch_bounds__(256) void wt_kernel(const float* __restrict__ W,
                                                 __bf16* __restrict__ Wt) {
    int idx = blockIdx.x * 256 + threadIdx.x;
    if (idx >= COUT * CIN) return;
    int e = idx / CIN;           // N index
    int c = idx - e * CIN;       // K index
    Wt[idx] = (__bf16)W[(size_t)c * COUT + e];
}

// ---------------------------------------------------------------------------
// Kernel 3: zero the two never-written parity sub-volumes:
//   which=0 -> (pd,ph,pw)=(0,1,1),  which=1 -> (1,1,0)
// ---------------------------------------------------------------------------
__global__ __launch_bounds__(256) void zero_kernel(float* __restrict__ out, int M) {
    int idx = blockIdx.x * 256 + threadIdx.x;
    if (idx >= M * 384) return;
    int site  = idx / 384;
    int r     = idx - site * 384;
    int which = r / 192;
    int c     = r - which * 192;
    int b = site >> 14, d = (site >> 10) & 15, h = (site >> 5) & 31, w = site & 31;
    int pd = which, ph = 1, pw = which ^ 1;
    size_t o = ((((size_t)b * 32 + 2 * d + pd) * 64 + (2 * h + ph)) * 64 +
                (2 * w + pw)) * 192 + c;
    out[o] = 0.f;
}

// ---------------------------------------------------------------------------
// Kernel 4: bf16 WMMA GEMM + parity scatter.
// Block = 256 threads = 8 waves; wave w owns the 16x16 tile at column
// n0_block + 16*w. grid.y in [0,9) covers only useful N blocks of 128:
// columns [0,384) U [768,1536).
// ---------------------------------------------------------------------------
__global__ __launch_bounds__(256) void gemm_kernel(const __bf16* __restrict__ A,
                                                   const __bf16* __restrict__ Bm,
                                                   float* __restrict__ out) {
    const int wave = threadIdx.x >> 5;
    const int lane = threadIdx.x & 31;
    const int lmod = lane & 15;
    const int lhi  = lane >> 4;

    const int m0 = blockIdx.x * 16;
    const int by = blockIdx.y;
    const int n0 = (by < 3 ? by * 128 : 768 + (by - 3) * 128) + wave * 16;

    // A fragment base: lane holds row m0+lmod; VGPR0-3 take K = k+8*lhi..+7,
    // VGPR4-7 take K = k+8*lhi+16..+23  (ISA 7.12.2, 16-bit A 16x32).
    const __bf16* arow = A  + (size_t)(m0 + lmod) * CIN + lhi * 8;
    // B fragment base: lane holds column n0+lmod; VGPR0-3 take K = k+8*lhi..+7,
    // VGPR4-7 take K = k+8*lhi+16..+23 (same order as A).
    const __bf16* brow = Bm + (size_t)(n0 + lmod) * CIN + lhi * 8;

    v8f acc = {};
#pragma unroll
    for (int k = 0; k < CIN; k += 32) {
        union { uint4 u[2]; v16bf v; } af, bf;
        af.u[0] = *(const uint4*)(arow + k);
        af.u[1] = *(const uint4*)(arow + k + 16);
        bf.u[0] = *(const uint4*)(brow + k);
        bf.u[1] = *(const uint4*)(brow + k + 16);
        acc = __builtin_amdgcn_wmma_f32_16x16x32_bf16(
            false, af.v, false, bf.v, (short)0, acc, false, false);
    }

    // Epilogue: C/D layout -> lane holds column N=lmod, rows M = r + 8*lhi.
    const int gn    = n0 + lmod;
    const int chunk = gn / 192;
    const int c     = gn - chunk * 192;
    // chunk -> parity index (pd*4+ph*2+pw), 3 bits each:
    // 0->0, 1->4, 4->5, 5->2, 6->1, 7->7  (2,3 never scheduled)
    const int p  = (0xE55020u >> (3 * chunk)) & 7;
    const int pd = p >> 2, ph = (p >> 1) & 1, pw = p & 1;

    union { v8f v; float f[8]; } cu;
    cu.v = acc;
#pragma unroll
    for (int r = 0; r < 8; ++r) {
        int gm = m0 + r + 8 * lhi;
        int b = gm >> 14, d = (gm >> 10) & 15, h = (gm >> 5) & 31, w = gm & 31;
        size_t o = ((((size_t)b * 32 + 2 * d + pd) * 64 + (2 * h + ph)) * 64 +
                    (2 * w + pw)) * 192 + c;
        __builtin_nontemporal_store(cu.f[r], &out[o]);
    }
}

// ---------------------------------------------------------------------------
extern "C" void kernel_launch(void* const* d_in, const int* in_sizes, int n_in,
                              void* d_out, int out_size, void* d_ws, size_t ws_size,
                              hipStream_t stream) {
    const float* x     = (const float*)d_in[0];
    const float* W     = (const float*)d_in[1];
    const float* gamma = (const float*)d_in[2];
    const float* beta  = (const float*)d_in[3];
    float* out = (float*)d_out;

    const int M = in_sizes[0] / CIN;  // 32768

    __bf16* xn = (__bf16*)d_ws;                 // M * 384 bf16  (~25 MB)
    __bf16* Wt = xn + (size_t)M * CIN;          // 1536 * 384 bf16 (~1.2 MB)

    ln_kernel<<<(M + 7) / 8, 256, 0, stream>>>(x, gamma, beta, xn, M);
    wt_kernel<<<(COUT * CIN + 255) / 256, 256, 0, stream>>>(W, Wt);
    zero_kernel<<<((size_t)M * 384 + 255) / 256, 256, 0, stream>>>(out, M);

    dim3 grid((M + 15) / 16, 9);
    gemm_kernel<<<grid, 256, 0, stream>>>(xn, Wt, out);
}
